// RecurrentGraphEmbeddingDS_3393024163881
// MI455X (gfx1250) — matrix operand order repaired and checked by reading, not compiled, not run on hardware
//
#include <hip/hip_runtime.h>

// ---------------------------------------------------------------------------
// RecurrentGraphEmbeddingDS for MI455X (gfx1250, wave32, WMMA bf16)
//
// Pipeline (batch arrays sorted -> CSR segment sum; no atomics):
//   Kprep: all weights -> transposed bf16 padded images in ws (one-time VALU)
//   K0:    starts[g] = lower_bound(batch, g) for batch1/batch2
//   K1:    nodes: h = MLP3(x1)               -> blocked bf16 tile buffer
//   KP:    pool hbuf by starts1              -> pooled1
//   K2:    graphs: u1 = MLP3(pooled1)
//   K3:    nodes: h = MLP3([x2 | u1[batch1]]) (split-K) -> hbuf
//   KP:    pool hbuf by starts2              -> pooled2
//   K4:    graphs: out = MLP3(MLP3(pooled2)) (6 fused layers, last 128->2)
//
// Notes:
//  - f32<->bf16 uses native v_cvt_pk_bf16_f32 (confirmed in codegen); these
//    VALU ops co-execute with the WMMA (TRANS) pipe, so conversions are free.
//  - The bf16f32-WMMA converter path is intentionally disabled: it occupies
//    the matrix pipe (the bottleneck) with zero-FLOP conversions.
// ---------------------------------------------------------------------------

typedef unsigned int u32;
typedef __attribute__((ext_vector_type(8)))  float   v8f;
typedef __attribute__((ext_vector_type(16))) __bf16  v16bf;
typedef __attribute__((ext_vector_type(2)))  __bf16  v2bf;

#define SW 136                 // image/LDS row stride in bf16 elems (16B rows)
#define IMG_U32 8704           // 128*SW/2 u32 per weight image
#define IMG_U4  2176           // uint4 per weight image
#define N_NODES  200000
#define N_GRAPHS 8192

union FragU { u32 u[8]; v16bf v; };

// f32 pair -> packed bf16 (lowers to one v_cvt_pk_bf16_f32)
__device__ __forceinline__ u32 pack2bf(float a, float b) {
  v2bf p;
  p[0] = (__bf16)a;
  p[1] = (__bf16)b;
  return __builtin_bit_cast(u32, p);
}

__device__ __forceinline__ unsigned short bf16u(float a) {
  __bf16 h = (__bf16)a;
  return __builtin_bit_cast(unsigned short, h);
}

// Build one 32-K B fragment (lane <-> column) from a row-major fp32 row
// pointer (p already offset by hi*16).
__device__ __forceinline__ void load_bfrag_f32(const float* p, u32 out[8]) {
  const float4* q = reinterpret_cast<const float4*>(p);
  float4 f0 = q[0], f1 = q[1], f2 = q[2], f3 = q[3];
  out[0] = pack2bf(f0.x, f0.y); out[1] = pack2bf(f0.z, f0.w);
  out[2] = pack2bf(f1.x, f1.y); out[3] = pack2bf(f1.z, f1.w);
  out[4] = pack2bf(f2.x, f2.y); out[5] = pack2bf(f2.z, f2.w);
  out[6] = pack2bf(f3.x, f3.y); out[7] = pack2bf(f3.z, f3.w);
}

__device__ __forceinline__ void build_bfrags(const float* row, int hi, u32 Bf[4][8]) {
  const float* rp = row + hi * 16;
#pragma unroll
  for (int kt = 0; kt < 4; ++kt) load_bfrag_f32(rp + kt * 32, Bf[kt]);
}

// Copy a pre-converted weight image (global, bf16, transposed+padded) to LDS.
__device__ __forceinline__ void stage_copy(uint4* WTv, float* BS, const u32* img,
                                           const float* b, int nb) {
  __syncthreads();
  const uint4* src = reinterpret_cast<const uint4*>(img);
  for (int i = threadIdx.x; i < IMG_U4; i += 256) WTv[i] = src[i];
  if (threadIdx.x < nb) BS[threadIdx.x] = b[threadIdx.x];
  __syncthreads();
}

// One 128->128 layer: accv[jt] += W^T_tile x Bfrag (32 v_wmma per call)
__device__ __forceinline__ void layer_accum(const unsigned short* WT, const u32 Bf[4][8],
                                            v8f accv[8], bool init, int colc, int hi) {
#pragma unroll
  for (int jt = 0; jt < 8; ++jt) {
    v8f acc = {0.f, 0.f, 0.f, 0.f, 0.f, 0.f, 0.f, 0.f};
    if (!init) acc = accv[jt];
#pragma unroll
    for (int kt = 0; kt < 4; ++kt) {
      FragU A, B;
      const unsigned short* row = WT + (jt * 16 + colc) * SW + kt * 32 + hi * 16;
      uint4 lo = *reinterpret_cast<const uint4*>(row);       // K = 16hi..16hi+7
      uint4 hh = *reinterpret_cast<const uint4*>(row + 8);   // K = 16hi+8..16hi+15
      A.u[0] = lo.x; A.u[1] = lo.y; A.u[2] = lo.z; A.u[3] = lo.w;
      A.u[4] = hh.x; A.u[5] = hh.y; A.u[6] = hh.z; A.u[7] = hh.w;
#pragma unroll
      for (int i = 0; i < 8; ++i) B.u[i] = Bf[kt][i];
      acc = __builtin_amdgcn_wmma_f32_16x16x32_bf16(false, A.v, false, B.v,
                                                    (short)0, acc, false, false);
    }
    accv[jt] = acc;
  }
}

__device__ __forceinline__ void bias_relu(const float* BS, v8f accv[8], bool relu, int hi) {
#pragma unroll
  for (int jt = 0; jt < 8; ++jt) {
    const float* bp = BS + jt * 16 + hi * 8;
    float4 b0 = *reinterpret_cast<const float4*>(bp);
    float4 b1 = *reinterpret_cast<const float4*>(bp + 4);
    v8f a = accv[jt];
    a[0] += b0.x; a[1] += b0.y; a[2] += b0.z; a[3] += b0.w;
    a[4] += b1.x; a[5] += b1.y; a[6] += b1.z; a[7] += b1.w;
    if (relu) {
#pragma unroll
      for (int r = 0; r < 8; ++r) a[r] = a[r] > 0.f ? a[r] : 0.f;
    }
    accv[jt] = a;
  }
}

// Convert one f32 D tile (8 VGPRs) to packed-bf16 P[0..3] (rows 2q,2q+1).
// Pure VALU (v_cvt_pk_bf16_f32) - co-executes with the WMMA pipe.
__device__ __forceinline__ void cvt_tile(const v8f& acc, u32 P[4]) {
#pragma unroll
  for (int q = 0; q < 4; ++q) P[q] = pack2bf(acc[2 * q], acc[2 * q + 1]);
}

// Register-only D-layout -> B-fragment conversion (half-wave ds_swizzle SWAPX16)
__device__ __forceinline__ void d_to_bfrag(const v8f accv[8], u32 Bf[4][8], int hi) {
#pragma unroll
  for (int t = 0; t < 4; ++t) {
    u32 P0[4], P1[4];
    cvt_tile(accv[2 * t], P0);
    cvt_tile(accv[2 * t + 1], P1);
#pragma unroll
    for (int q = 0; q < 4; ++q) {
      u32 s0 = (u32)__builtin_amdgcn_ds_swizzle((int)P0[q], 0x401F); // SWAPX16
      u32 s1 = (u32)__builtin_amdgcn_ds_swizzle((int)P1[q], 0x401F);
      Bf[t][q]     = hi ? s1 : P0[q];
      Bf[t][q + 4] = hi ? P1[q] : s0;
    }
  }
}

// Store one wave's 16x128 bf16 tile to blocked buffer: coalesced b128 stores.
__device__ __forceinline__ void store_htile(u32* hbuf, int tile, int lane,
                                            const u32 Bf[4][8]) {
  uint4* dst = reinterpret_cast<uint4*>(hbuf) + (size_t)tile * 256 + lane;
#pragma unroll
  for (int c = 0; c < 8; ++c) {
    uint4 d;
    d.x = Bf[c >> 1][(c & 1) * 4 + 0];
    d.y = Bf[c >> 1][(c & 1) * 4 + 1];
    d.z = Bf[c >> 1][(c & 1) * 4 + 2];
    d.w = Bf[c >> 1][(c & 1) * 4 + 3];
    dst[c * 32] = d;
  }
}

// ---------------------------------------------------------------------------
// Kprep: one-time weight conversion into transposed bf16 padded images.
// Images 0..14: [128 x 128]; image 15: final [128 x 2] into 16 rows (zero pad).
// ---------------------------------------------------------------------------
struct PrepArgs { const float* w[16]; };

__global__ __launch_bounds__(256) void k_prep(PrepArgs a, u32* __restrict__ wimg) {
  const int b = blockIdx.x;
  unsigned short* dst = reinterpret_cast<unsigned short*>(wimg + (size_t)b * IMG_U32);
  const float* W = a.w[b];
  if (b < 15) {
    for (int idx = threadIdx.x; idx < 128 * 128; idx += 256) {
      int k = idx >> 7, j = idx & 127;
      dst[j * SW + k] = bf16u(W[idx]);
    }
  } else {
    for (int idx = threadIdx.x; idx < 16 * 128; idx += 256) {
      int k = idx & 127, j = idx >> 7;
      dst[j * SW + k] = bf16u((j < 2) ? W[k * 2 + j] : 0.f);
    }
  }
}

// ---------------------------------------------------------------------------
// K0: CSR starts via binary search over sorted batch arrays
// ---------------------------------------------------------------------------
__global__ __launch_bounds__(256) void k_starts(const int* __restrict__ b1,
                                                const int* __restrict__ b2,
                                                int* __restrict__ s1,
                                                int* __restrict__ s2, int n) {
  int g = blockIdx.x * 256 + threadIdx.x;
  if (g > N_GRAPHS) return;
  int lo = 0, hi = n;
  while (lo < hi) { int mid = (lo + hi) >> 1; if (b1[mid] < g) lo = mid + 1; else hi = mid; }
  s1[g] = lo;
  lo = 0; hi = n;
  while (lo < hi) { int mid = (lo + hi) >> 1; if (b2[mid] < g) lo = mid + 1; else hi = mid; }
  s2[g] = lo;
}

// ---------------------------------------------------------------------------
// KP: segment-sum pooling, one wave per graph.
// ---------------------------------------------------------------------------
__global__ __launch_bounds__(256) void k_pool(const u32* __restrict__ hbuf,
                                              const int* __restrict__ starts,
                                              float* __restrict__ pooled) {
  const int lane = threadIdx.x & 31, wv = threadIdx.x >> 5;
  const int hi = (lane >> 4) & 1, colc = lane & 15;
  const int g = blockIdx.x * 8 + wv;
  const int s = starts[g], e = starts[g + 1];
  float acc[64];
#pragma unroll
  for (int i = 0; i < 64; ++i) acc[i] = 0.f;
  const int t0 = s >> 4, t1 = (e + 15) >> 4;
  for (int t = t0; t < t1; ++t) {
    const int n = t * 16 + colc;
    if (n >= s && n < e) {
      const uint4* p = reinterpret_cast<const uint4*>(hbuf) + (size_t)t * 256 + lane;
#pragma unroll
      for (int c = 0; c < 8; ++c) {
        uint4 d = p[c * 32];
        u32 w0 = d.x, w1 = d.y, w2 = d.z, w3 = d.w;
        acc[c * 8 + 0] += __uint_as_float(w0 << 16);
        acc[c * 8 + 1] += __uint_as_float(w0 & 0xFFFF0000u);
        acc[c * 8 + 2] += __uint_as_float(w1 << 16);
        acc[c * 8 + 3] += __uint_as_float(w1 & 0xFFFF0000u);
        acc[c * 8 + 4] += __uint_as_float(w2 << 16);
        acc[c * 8 + 5] += __uint_as_float(w2 & 0xFFFF0000u);
        acc[c * 8 + 6] += __uint_as_float(w3 << 16);
        acc[c * 8 + 7] += __uint_as_float(w3 & 0xFFFF0000u);
      }
    }
  }
#pragma unroll
  for (int i = 0; i < 64; ++i) {
    float v = acc[i];
    v += __int_as_float(__builtin_amdgcn_ds_swizzle(__float_as_int(v), 0x041F));
    v += __int_as_float(__builtin_amdgcn_ds_swizzle(__float_as_int(v), 0x081F));
    v += __int_as_float(__builtin_amdgcn_ds_swizzle(__float_as_int(v), 0x101F));
    v += __int_as_float(__builtin_amdgcn_ds_swizzle(__float_as_int(v), 0x201F));
    acc[i] = v;
  }
  if (colc == 0) {
    float* dst = pooled + (size_t)g * 128 + hi * 16;
#pragma unroll
    for (int c = 0; c < 8; ++c) {
      const int kt = c >> 1, vb = (c & 1) * 4;
#pragma unroll
      for (int q = 0; q < 4; ++q) {
        float2 o;
        o.x = acc[c * 8 + 2 * q];
        o.y = acc[c * 8 + 2 * q + 1];
        *reinterpret_cast<float2*>(dst + kt * 32 + 2 * (vb + q)) = o;
      }
    }
  }
}

// ---------------------------------------------------------------------------
// K1: DS1 node MLP (x1 -> h tiles)   [images 0,1,2]
// ---------------------------------------------------------------------------
__global__ __launch_bounds__(256) void k_ds1_nodes(
    const float* __restrict__ x, const u32* __restrict__ wimg,
    const float* b0, const float* b1, const float* b2,
    u32* __restrict__ hbuf, int nNodes) {
  __shared__ uint4 WTv[IMG_U4];
  __shared__ float BS[128];
  unsigned short* WT = reinterpret_cast<unsigned short*>(WTv);
  const int tid = threadIdx.x, lane = tid & 31, wv = tid >> 5;
  const int hi = (lane >> 4) & 1, colc = lane & 15;
  const int base = blockIdx.x * 128 + wv * 16;
  const int node = base + colc;
  const int nodeC = node < nNodes ? node : nNodes - 1;

  u32 Bf[4][8];
  build_bfrags(x + (size_t)nodeC * 128, hi, Bf);

  v8f accv[8];
  stage_copy(WTv, BS, wimg + 0 * IMG_U32, b0, 128);
  layer_accum(WT, Bf, accv, true, colc, hi);
  bias_relu(BS, accv, true, hi);
  d_to_bfrag(accv, Bf, hi);

  stage_copy(WTv, BS, wimg + 1 * IMG_U32, b1, 128);
  layer_accum(WT, Bf, accv, true, colc, hi);
  bias_relu(BS, accv, true, hi);
  d_to_bfrag(accv, Bf, hi);

  stage_copy(WTv, BS, wimg + 2 * IMG_U32, b2, 128);
  layer_accum(WT, Bf, accv, true, colc, hi);
  bias_relu(BS, accv, false, hi);
  d_to_bfrag(accv, Bf, hi);

  if (base < nNodes) store_htile(hbuf, base >> 4, lane, Bf);
}

// ---------------------------------------------------------------------------
// K2: DS1 out MLP (pooled1 -> u1)    [images 3,4,5]
// ---------------------------------------------------------------------------
__global__ __launch_bounds__(256) void k_ds1_out(
    const float* __restrict__ pooled, const u32* __restrict__ wimg,
    const float* b0, const float* b1, const float* b2, float* __restrict__ u1) {
  __shared__ uint4 WTv[IMG_U4];
  __shared__ float BS[128];
  unsigned short* WT = reinterpret_cast<unsigned short*>(WTv);
  const int tid = threadIdx.x, lane = tid & 31, wv = tid >> 5;
  const int hi = (lane >> 4) & 1, colc = lane & 15;
  const int grow = blockIdx.x * 128 + wv * 16 + colc;

  u32 Bf[4][8];
  build_bfrags(pooled + (size_t)grow * 128, hi, Bf);

  v8f accv[8];
  stage_copy(WTv, BS, wimg + 3 * IMG_U32, b0, 128);
  layer_accum(WT, Bf, accv, true, colc, hi);
  bias_relu(BS, accv, true, hi);
  d_to_bfrag(accv, Bf, hi);

  stage_copy(WTv, BS, wimg + 4 * IMG_U32, b1, 128);
  layer_accum(WT, Bf, accv, true, colc, hi);
  bias_relu(BS, accv, true, hi);
  d_to_bfrag(accv, Bf, hi);

  stage_copy(WTv, BS, wimg + 5 * IMG_U32, b2, 128);
  layer_accum(WT, Bf, accv, true, colc, hi);
  bias_relu(BS, accv, false, hi);

  float* dst = u1 + (size_t)grow * 128 + hi * 8;
#pragma unroll
  for (int jt = 0; jt < 8; ++jt)
#pragma unroll
    for (int r = 0; r < 8; ++r) dst[jt * 16 + r] = accv[jt][r];
}

// ---------------------------------------------------------------------------
// K3: DS2 node MLP on concat([x2, u1[batch1]]) via split-K  [images 6,7,8,9]
// ---------------------------------------------------------------------------
__global__ __launch_bounds__(256) void k_ds2_nodes(
    const float* __restrict__ x2, const int* __restrict__ batch1,
    const float* __restrict__ u1, const u32* __restrict__ wimg,
    const float* b0, const float* b1, const float* b2,
    u32* __restrict__ hbuf, int nNodes) {
  __shared__ uint4 WTv[IMG_U4];
  __shared__ float BS[128];
  unsigned short* WT = reinterpret_cast<unsigned short*>(WTv);
  const int tid = threadIdx.x, lane = tid & 31, wv = tid >> 5;
  const int hi = (lane >> 4) & 1, colc = lane & 15;
  const int base = blockIdx.x * 128 + wv * 16;
  const int node = base + colc;
  const int nodeC = node < nNodes ? node : nNodes - 1;

  v8f accv[8];
  u32 Bf[4][8];

  build_bfrags(x2 + (size_t)nodeC * 128, hi, Bf);
  stage_copy(WTv, BS, wimg + 6 * IMG_U32, b0, 128);   // W rows 0..127
  layer_accum(WT, Bf, accv, true, colc, hi);

  const int g1 = batch1[nodeC];
  build_bfrags(u1 + (size_t)g1 * 128, hi, Bf);
  stage_copy(WTv, BS, wimg + 7 * IMG_U32, b0, 128);   // W rows 128..255
  layer_accum(WT, Bf, accv, false, colc, hi);
  bias_relu(BS, accv, true, hi);
  d_to_bfrag(accv, Bf, hi);

  stage_copy(WTv, BS, wimg + 8 * IMG_U32, b1, 128);
  layer_accum(WT, Bf, accv, true, colc, hi);
  bias_relu(BS, accv, true, hi);
  d_to_bfrag(accv, Bf, hi);

  stage_copy(WTv, BS, wimg + 9 * IMG_U32, b2, 128);
  layer_accum(WT, Bf, accv, true, colc, hi);
  bias_relu(BS, accv, false, hi);
  d_to_bfrag(accv, Bf, hi);

  if (base < nNodes) store_htile(hbuf, base >> 4, lane, Bf);
}

// ---------------------------------------------------------------------------
// K4: fused 6-layer graph head: pooled2 -> out[8192,2]   [images 10..15]
// ---------------------------------------------------------------------------
__global__ __launch_bounds__(256) void k_graph_head(
    const float* __restrict__ pooled, const u32* __restrict__ wimg,
    const float* o0b, const float* o1b, const float* o2b,
    const float* m0b, const float* m1b, const float* m2b,
    float* __restrict__ out) {
  __shared__ uint4 WTv[IMG_U4];
  __shared__ float BS[128];
  unsigned short* WT = reinterpret_cast<unsigned short*>(WTv);
  const int tid = threadIdx.x, lane = tid & 31, wv = tid >> 5;
  const int hi = (lane >> 4) & 1, colc = lane & 15;
  const int grow = blockIdx.x * 128 + wv * 16 + colc;

  u32 Bf[4][8];
  build_bfrags(pooled + (size_t)grow * 128, hi, Bf);

  v8f accv[8];
  stage_copy(WTv, BS, wimg + 10 * IMG_U32, o0b, 128);
  layer_accum(WT, Bf, accv, true, colc, hi);
  bias_relu(BS, accv, true, hi);
  d_to_bfrag(accv, Bf, hi);

  stage_copy(WTv, BS, wimg + 11 * IMG_U32, o1b, 128);
  layer_accum(WT, Bf, accv, true, colc, hi);
  bias_relu(BS, accv, true, hi);
  d_to_bfrag(accv, Bf, hi);

  stage_copy(WTv, BS, wimg + 12 * IMG_U32, o2b, 128);
  layer_accum(WT, Bf, accv, true, colc, hi);
  bias_relu(BS, accv, false, hi);            // u2 (linear)
  d_to_bfrag(accv, Bf, hi);

  stage_copy(WTv, BS, wimg + 13 * IMG_U32, m0b, 128);
  layer_accum(WT, Bf, accv, true, colc, hi);
  bias_relu(BS, accv, true, hi);
  d_to_bfrag(accv, Bf, hi);

  stage_copy(WTv, BS, wimg + 14 * IMG_U32, m1b, 128);
  layer_accum(WT, Bf, accv, true, colc, hi);
  bias_relu(BS, accv, true, hi);
  d_to_bfrag(accv, Bf, hi);

  // Final 128 -> 2 layer
  stage_copy(WTv, BS, wimg + 15 * IMG_U32, m2b, 2);
  v8f acc = {0.f, 0.f, 0.f, 0.f, 0.f, 0.f, 0.f, 0.f};
#pragma unroll
  for (int kt = 0; kt < 4; ++kt) {
    FragU A, B;
    const unsigned short* row = WT + colc * SW + kt * 32 + hi * 16;
    uint4 lo = *reinterpret_cast<const uint4*>(row);
    uint4 hh = *reinterpret_cast<const uint4*>(row + 8);
    A.u[0] = lo.x; A.u[1] = lo.y; A.u[2] = lo.z; A.u[3] = lo.w;
    A.u[4] = hh.x; A.u[5] = hh.y; A.u[6] = hh.z; A.u[7] = hh.w;
#pragma unroll
    for (int i = 0; i < 8; ++i) B.u[i] = Bf[kt][i];
    acc = __builtin_amdgcn_wmma_f32_16x16x32_bf16(false, A.v, false, B.v,
                                                  (short)0, acc, false, false);
  }
  if (hi == 0) {
    float2 o;
    o.x = acc[0] + BS[0];
    o.y = acc[1] + BS[1];
    reinterpret_cast<float2*>(out)[grow] = o;
  }
}

// ---------------------------------------------------------------------------
extern "C" void kernel_launch(void* const* d_in, const int* in_sizes, int n_in,
                              void* d_out, int out_size, void* d_ws, size_t ws_size,
                              hipStream_t stream) {
  const float* x1     = (const float*)d_in[0];
  const int*   batch1 = (const int*)d_in[1];
  const float* x2     = (const float*)d_in[2];
  const int*   batch2 = (const int*)d_in[3];
  const float* e1W0 = (const float*)d_in[4];  const float* e1b0 = (const float*)d_in[5];
  const float* e1W1 = (const float*)d_in[6];  const float* e1b1 = (const float*)d_in[7];
  const float* e1W2 = (const float*)d_in[8];  const float* e1b2 = (const float*)d_in[9];
  const float* o1W0 = (const float*)d_in[10]; const float* o1b0 = (const float*)d_in[11];
  const float* o1W1 = (const float*)d_in[12]; const float* o1b1 = (const float*)d_in[13];
  const float* o1W2 = (const float*)d_in[14]; const float* o1b2 = (const float*)d_in[15];
  const float* e2W0 = (const float*)d_in[16]; const float* e2b0 = (const float*)d_in[17];
  const float* e2W1 = (const float*)d_in[18]; const float* e2b1 = (const float*)d_in[19];
  const float* e2W2 = (const float*)d_in[20]; const float* e2b2 = (const float*)d_in[21];
  const float* o2W0 = (const float*)d_in[22]; const float* o2b0 = (const float*)d_in[23];
  const float* o2W1 = (const float*)d_in[24]; const float* o2b1 = (const float*)d_in[25];
  const float* o2W2 = (const float*)d_in[26]; const float* o2b2 = (const float*)d_in[27];
  const float* mW0 = (const float*)d_in[28];  const float* mb0 = (const float*)d_in[29];
  const float* mW1 = (const float*)d_in[30];  const float* mb1 = (const float*)d_in[31];
  const float* mW2 = (const float*)d_in[32];  const float* mb2 = (const float*)d_in[33];

  char* ws = (char*)d_ws;
  float* pooled1 = (float*)(ws);                                //  4 MB
  float* u1      = (float*)(ws + (4u << 20));                   //  4 MB
  float* pooled2 = (float*)(ws + (8u << 20));                   //  4 MB
  int*   starts1 = (int*)(ws + (12u << 20));                    // 64 KB
  int*   starts2 = (int*)(ws + (12u << 20) + (64u << 10));      // 64 KB
  u32*   wimg    = (u32*)(ws + (12u << 20) + (128u << 10));     // 557 KB
  u32*   hbuf    = (u32*)(ws + (14u << 20));                    // 51.2 MB

  PrepArgs pa;
  pa.w[0] = e1W0; pa.w[1] = e1W1; pa.w[2] = e1W2;
  pa.w[3] = o1W0; pa.w[4] = o1W1; pa.w[5] = o1W2;
  pa.w[6] = e2W0; pa.w[7] = e2W0 + 128 * 128; pa.w[8] = e2W1; pa.w[9] = e2W2;
  pa.w[10] = o2W0; pa.w[11] = o2W1; pa.w[12] = o2W2;
  pa.w[13] = mW0; pa.w[14] = mW1; pa.w[15] = mW2;

  const int nodeBlocks  = (N_NODES + 127) / 128;   // 1563
  const int graphBlocks = N_GRAPHS / 128;          // 64
  const int poolBlocks  = N_GRAPHS / 8;            // 1024

  k_prep<<<16, 256, 0, stream>>>(pa, wimg);
  k_starts<<<(N_GRAPHS + 256) / 256, 256, 0, stream>>>(batch1, batch2, starts1,
                                                       starts2, N_NODES);
  k_ds1_nodes<<<nodeBlocks, 256, 0, stream>>>(x1, wimg, e1b0, e1b1, e1b2,
                                              hbuf, N_NODES);
  k_pool<<<poolBlocks, 256, 0, stream>>>(hbuf, starts1, pooled1);
  k_ds1_out<<<graphBlocks, 256, 0, stream>>>(pooled1, wimg, o1b0, o1b1, o1b2, u1);
  k_ds2_nodes<<<nodeBlocks, 256, 0, stream>>>(x2, batch1, u1, wimg,
                                              e2b0, e2b1, e2b2, hbuf, N_NODES);
  k_pool<<<poolBlocks, 256, 0, stream>>>(hbuf, starts2, pooled2);
  k_graph_head<<<graphBlocks, 256, 0, stream>>>(pooled2, wimg, o2b0, o2b1, o2b2,
                                                mb0, mb1, mb2, (float*)d_out);
}
